// QuantumLayer_50087908606673
// MI455X (gfx1250) — compile-verified, matrix operand order checked by reading, not run on hardware
//
#include <hip/hip_runtime.h>

// MI455X / gfx1250 (wave32, WMMA). Quantum layer recast as:
//   out = |A @ U^T|^2 @ SIGNS,  A real product-state matrix, U fixed unitary.
// Kernel 1: build U^T (half) by simulating 1024 basis columns in LDS.
// Kernel 2: fused f16 WMMA GEMM (Re+Im) + |.|^2 + sign-reduction WMMA.

#define NQ    10
#define DIM   1024
#define BM    16        // sample rows per workgroup in GEMM

typedef _Float16 v16h __attribute__((ext_vector_type(16)));
typedef _Float16 v8h  __attribute__((ext_vector_type(8)));
typedef float    v8f  __attribute__((ext_vector_type(8)));

__device__ __forceinline__ v16h ld_frag(const _Float16* p0, const _Float16* p1) {
  v8h a = *(const v8h*)p0;      // K = base .. base+7
  v8h b = *(const v8h*)p1;      // K = base+16 .. base+23
  v16h r;
  #pragma unroll
  for (int t = 0; t < 8; ++t) { r[t] = a[t]; r[t + 8] = b[t]; }
  return r;
}

// ---------------------------------------------------------------------------
// Kernel 1: simulate basis column j -> U[:,j]; store W^T[i][j] = U[i,j] (f16)
// ---------------------------------------------------------------------------
__global__ __launch_bounds__(256)
void build_unitary(const float* __restrict__ w,
                   _Float16* __restrict__ WrT, _Float16* __restrict__ WiT) {
  __shared__ float re[DIM];
  __shared__ float im[DIM];
  const int tid = threadIdx.x;
  const int col = blockIdx.x;

  for (int i = tid; i < DIM; i += 256) { re[i] = (i == col) ? 1.f : 0.f; im[i] = 0.f; }
  __syncthreads();

  int wi = 0;
  for (int layer = 0; layer < 2; ++layer) {
    // Rot(phi,theta,omega) = RZ(omega) RY(theta) RZ(phi) on qubit q
    for (int q = 0; q < NQ; ++q) {
      float phi = w[wi], th = w[wi + 1], om = w[wi + 2]; wi += 3;
      float al = 0.5f * (phi + om), dl = 0.5f * (phi - om);
      float c = cosf(0.5f * th), s = sinf(0.5f * th);
      float ca = cosf(al), sa = sinf(al), cd = cosf(dl), sd = sinf(dl);
      float u00r =  c * ca, u00i = -c * sa;
      float u01r = -s * cd, u01i = -s * sd;
      float u10r =  s * cd, u10i = -s * sd;
      float u11r =  c * ca, u11i =  c * sa;
      const int bp = (NQ - 1) - q, stride = 1 << bp;   // wire 0 = MSB
      #pragma unroll 2
      for (int p = tid; p < DIM / 2; p += 256) {
        int i0 = ((p >> bp) << (bp + 1)) | (p & (stride - 1));
        int i1 = i0 | stride;
        float s0r = re[i0], s0i = im[i0], s1r = re[i1], s1i = im[i1];
        re[i0] = u00r * s0r - u00i * s0i + u01r * s1r - u01i * s1i;
        im[i0] = u00r * s0i + u00i * s0r + u01r * s1i + u01i * s1r;
        re[i1] = u10r * s0r - u10i * s0i + u11r * s1r - u11i * s1i;
        im[i1] = u10r * s0i + u10i * s0r + u11r * s1i + u11i * s1r;
      }
      __syncthreads();
    }
    // CRY(theta) on ring edges (e, e+1)
    for (int e = 0; e < NQ; ++e) {
      float th = w[wi]; wi += 1;
      float c = cosf(0.5f * th), s = sinf(0.5f * th);
      const int u = e, v = (e + 1) % NQ;
      const int bc = (NQ - 1) - u, bt = (NQ - 1) - v;
      const int l = bc < bt ? bc : bt, h = bc < bt ? bt : bc;
      {
        int p = tid;                             // DIM/4 = 256 pairs
        int x = ((p >> l) << (l + 1)) | (p & ((1 << l) - 1));
        x = ((x >> h) << (h + 1)) | (x & ((1 << h) - 1));
        int i0 = x | (1 << bc);                  // control bit set
        int i1 = i0 | (1 << bt);
        float s0r = re[i0], s0i = im[i0], s1r = re[i1], s1i = im[i1];
        re[i0] = c * s0r - s * s1r;  im[i0] = c * s0i - s * s1i;
        re[i1] = s * s0r + c * s1r;  im[i1] = s * s0i + c * s1i;
      }
      __syncthreads();
    }
  }
  for (int i = tid; i < DIM; i += 256) {
    WrT[(size_t)i * DIM + col] = (_Float16)re[i];
    WiT[(size_t)i * DIM + col] = (_Float16)im[i];
  }
}

// ---------------------------------------------------------------------------
// Kernel 2: per 16-row block: A on the fly in LDS, WMMA GEMM over all 1024
// output amplitudes, fused probability + PauliZ sign reduction (also WMMA).
// ---------------------------------------------------------------------------
__global__ __launch_bounds__(256)
void qnn_wmma(const float* __restrict__ inputs,
              const _Float16* __restrict__ WrT,
              const _Float16* __restrict__ WiT,
              float* __restrict__ out) {
  __shared__ __align__(16) _Float16 A_sh[BM][DIM];     // 32 KB
  __shared__ float trig[BM][NQ][2];
  __shared__ __align__(16) _Float16 p_sh[8][16][32];   // 8 KB, per-wave scratch
  __shared__ float out_sh[16][16];

  const int tid  = threadIdx.x;
  const int row0 = blockIdx.x * BM;

  out_sh[tid >> 4][tid & 15] = 0.f;
  if (tid < BM * NQ) {
    int m = tid / NQ, q = tid % NQ;
    float x = inputs[(size_t)(row0 + m) * NQ + q];
    x = fminf(fmaxf(x, 0.f), 1.f);
    float ha = 0.5f * 3.14159265358979323846f * x;   // half of pi*x
    trig[m][q][0] = cosf(ha);
    trig[m][q][1] = sinf(ha);
  }
  __syncthreads();

  // A[m][j] = prod_q (bit ? sin : cos); wire 0 is MSB of j
  for (int idx = tid; idx < BM * DIM; idx += 256) {
    int m = idx >> 10, j = idx & (DIM - 1);
    float prod = 1.f;
    #pragma unroll
    for (int q = 0; q < NQ; ++q)
      prod *= trig[m][q][(j >> ((NQ - 1) - q)) & 1];
    A_sh[m][j] = (_Float16)prod;
  }
  __syncthreads();

  const int wave = tid >> 5, lane = tid & 31;
  const int lo = lane & 15, hi = lane >> 4;

  v8f oacc = {};                                  // 16 rows x 16 (10 used) q
  for (int t = 0; t < 4; ++t) {                   // 8 waves x 4 = 32 col supertiles
    const int ns = (wave + 8 * t) * 32;           // 32 output amplitudes
    v8f cr0 = {}, ci0 = {}, cr1 = {}, ci1 = {};
    for (int k0 = 0; k0 < DIM; k0 += 32) {
      v16h a = ld_frag(&A_sh[lo][k0 + hi * 8], &A_sh[lo][k0 + 16 + hi * 8]);
      const _Float16* wr0 = WrT + (size_t)(ns + lo)      * DIM + k0 + hi * 8;
      const _Float16* wi0 = WiT + (size_t)(ns + lo)      * DIM + k0 + hi * 8;
      const _Float16* wr1 = WrT + (size_t)(ns + 16 + lo) * DIM + k0 + hi * 8;
      const _Float16* wi1 = WiT + (size_t)(ns + 16 + lo) * DIM + k0 + hi * 8;
      v16h br0 = ld_frag(wr0, wr0 + 16);
      v16h bi0 = ld_frag(wi0, wi0 + 16);
      v16h br1 = ld_frag(wr1, wr1 + 16);
      v16h bi1 = ld_frag(wi1, wi1 + 16);
      cr0 = __builtin_amdgcn_wmma_f32_16x16x32_f16(false, a, false, br0, (short)0, cr0, false, false);
      ci0 = __builtin_amdgcn_wmma_f32_16x16x32_f16(false, a, false, bi0, (short)0, ci0, false, false);
      cr1 = __builtin_amdgcn_wmma_f32_16x16x32_f16(false, a, false, br1, (short)0, cr1, false, false);
      ci1 = __builtin_amdgcn_wmma_f32_16x16x32_f16(false, a, false, bi1, (short)0, ci1, false, false);
    }
    // probabilities for this 16x32 tile -> per-wave LDS (C-layout -> A-layout)
    #pragma unroll
    for (int r = 0; r < 8; ++r) {
      p_sh[wave][r + 8 * hi][lo]      = (_Float16)(cr0[r] * cr0[r] + ci0[r] * ci0[r]);
      p_sh[wave][r + 8 * hi][16 + lo] = (_Float16)(cr1[r] * cr1[r] + ci1[r] * ci1[r]);
    }
    v16h pf = ld_frag(&p_sh[wave][lo][hi * 8], &p_sh[wave][lo][16 + hi * 8]);
    // sign B-fragment: S[kr][q] = 1 - 2*bit_{q}(ns+kr), cols >= NQ zeroed
    v16h sf;
    #pragma unroll
    for (int tt = 0; tt < 16; ++tt) {
      int kr = (tt < 8 ? 0 : 16) + hi * 8 + (tt & 7);
      int i  = ns + kr;
      float sv = (lo < NQ) ? (1.f - 2.f * (float)((i >> ((NQ - 1) - lo)) & 1)) : 0.f;
      sf[tt] = (_Float16)sv;
    }
    oacc = __builtin_amdgcn_wmma_f32_16x16x32_f16(false, pf, false, sf, (short)0, oacc, false, false);
  }

  if (lo < NQ) {
    #pragma unroll
    for (int r = 0; r < 8; ++r)
      atomicAdd(&out_sh[r + 8 * hi][lo], oacc[r]);   // ds_add_f32, 8 col-waves
  }
  __syncthreads();
  if (tid < BM * NQ) {
    int m = tid / NQ, q = tid % NQ;
    out[(size_t)(row0 + m) * NQ + q] = out_sh[m][q];
  }
}

// ---------------------------------------------------------------------------
extern "C" void kernel_launch(void* const* d_in, const int* in_sizes, int n_in,
                              void* d_out, int out_size, void* d_ws, size_t ws_size,
                              hipStream_t stream) {
  const float* inputs  = (const float*)d_in[0];   // (B, NQ) f32
  const float* weights = (const float*)d_in[1];   // (80,)   f32
  float* out = (float*)d_out;                     // (B, NQ) f32

  _Float16* WrT = (_Float16*)d_ws;                // DIM*DIM halves (2 MB)
  _Float16* WiT = WrT + (size_t)DIM * DIM;        // DIM*DIM halves (2 MB)

  const int B = in_sizes[0] / NQ;                 // 32768

  build_unitary<<<DIM, 256, 0, stream>>>(weights, WrT, WiT);
  qnn_wmma<<<B / BM, 256, 0, stream>>>(inputs, WrT, WiT, out);
}
